// Attention_25967372271784
// MI455X (gfx1250) — matrix operand order repaired and checked by reading, not compiled, not run on hardware
//
#include <hip/hip_runtime.h>

#define NUM_HEADS 16
#define EMBED     1024
#define HEAD_DIM  64
#define SEQ       2048
#define BATCH     2
#define ATT_SCALE (-32.0f)   // HEAD_DIM * -0.5, faithful to reference

// padded LDS row strides (elements) -> bank-conflict-free ds_load_b128
#define VROW   72    // 64 + 8  (32 dwords + 4 pad dwords)
#define VTROW  136   // 128 + 8 (64 dwords + 4 pad dwords)
#define PROW   136

typedef __attribute__((ext_vector_type(16))) unsigned short v16u;
typedef __attribute__((ext_vector_type(8)))  unsigned short v8u;
typedef __attribute__((ext_vector_type(16))) __bf16        v16bf;
typedef __attribute__((ext_vector_type(8)))  float         v8f;
typedef __attribute__((ext_vector_type(4)))  unsigned int  u32x4;
typedef __attribute__((ext_vector_type(4)))  int           i32x4;
typedef __attribute__((ext_vector_type(8)))  int           i32x8;

#if defined(__has_builtin)
#  if __has_builtin(__builtin_amdgcn_tensor_load_to_lds) && \
      __has_builtin(__builtin_amdgcn_s_wait_tensorcnt)
#    define HAVE_TDM 1
#  endif
#endif
#ifndef HAVE_TDM
#  define HAVE_TDM 0
#endif

static __device__ __forceinline__ unsigned short f2bf(float f) {
  unsigned u = __builtin_bit_cast(unsigned, f);
  u += 0x7FFFu + ((u >> 16) & 1u);          // round-to-nearest-even
  return (unsigned short)(u >> 16);
}

static __device__ __forceinline__ v8f vzero8() {
  v8f z;
  #pragma unroll
  for (int i = 0; i < 8; ++i) z[i] = 0.0f;
  return z;
}

// Concatenate two 8-element chunks (A-operand halves, or a 16-elem 16B-aligned pair).
static __device__ __forceinline__ v16bf mergeA(v8u lo, v8u hi) {
  v16u t;
  #pragma unroll
  for (int i = 0; i < 8; ++i) { t[i] = lo[i]; t[i + 8] = hi[i]; }
  return __builtin_bit_cast(v16bf, t);
}
static __device__ __forceinline__ v16bf asbf(v16u t) {
  return __builtin_bit_cast(v16bf, t);
}
// 16 contiguous bf16 from a (>=16B aligned) pointer, via two v8u loads.
static __device__ __forceinline__ v16bf ld16(const unsigned short* p) {
  return mergeA(*(const v8u*)p, *(const v8u*)(p + 8));
}

// ---------------------------------------------------------------- TDM issue
// 2-D tile DMA: tensor (tensor_d0 x tensor_d1) elements, 2B each, row stride
// stride0 elements; tile (tile_d0 x tile_d1) starting at gaddr -> LDS lds_byte,
// with LDS padding pad_amount(=4) dwords every 2^(pad_ivl_code+1) dwords.
static __device__ __forceinline__ void tdm_load_2d(unsigned lds_byte, const void* gaddr,
                                                   unsigned tensor_d0, unsigned tensor_d1,
                                                   unsigned tile_d0, unsigned tile_d1,
                                                   unsigned stride0, unsigned pad_ivl_code) {
#if HAVE_TDM
  unsigned long long ga = (unsigned long long)(uintptr_t)gaddr;
  u32x4 g0;
  g0[0] = 1u;                                              // count=1, user desc
  g0[1] = lds_byte;                                        // lds_addr
  g0[2] = (unsigned)(ga & 0xFFFFFFFFull);                  // global_addr[31:0]
  g0[3] = (unsigned)((ga >> 32) & 0x1FFFFFFull)            // global_addr[56:32]
        | (2u << 30);                                      // type=2 ("image")
  i32x8 g1;
  g1[0] = (int)((1u << 16)                                 // data_size = 2 bytes
        | (1u << 20)                                       // pad_enable
        | (pad_ivl_code << 22)                             // pad_interval
        | (3u << 25));                                     // pad_amount = 4 dwords
  g1[1] = (int)((tensor_d0 & 0xFFFFu) << 16);              // tensor_dim0[15:0]
  g1[2] = (int)(((tensor_d0 >> 16) & 0xFFFFu) | ((tensor_d1 & 0xFFFFu) << 16));
  g1[3] = (int)(((tensor_d1 >> 16) & 0xFFFFu) | ((tile_d0 & 0xFFFFu) << 16));
  g1[4] = (int)(tile_d1 & 0xFFFFu);                        // tile_dim1 (tile_dim2=0)
  g1[5] = (int)stride0;                                    // tensor_dim0_stride[31:0]
  g1[6] = 0;
  g1[7] = 0;
  i32x4 z4; z4[0] = z4[1] = z4[2] = z4[3] = 0;
#if __clang_major__ >= 23
  i32x8 z8;
  #pragma unroll
  for (int i = 0; i < 8; ++i) z8[i] = 0;
  __builtin_amdgcn_tensor_load_to_lds(g0, g1, z4, z4, z8, 0);
#else
  __builtin_amdgcn_tensor_load_to_lds(g0, g1, z4, z4, 0);
#endif
#else
  (void)lds_byte; (void)gaddr; (void)tensor_d0; (void)tensor_d1;
  (void)tile_d0; (void)tile_d1; (void)stride0; (void)pad_ivl_code;
#endif
}

// ---------------------------------------------------------------- convert
__global__ void cvt_f32_bf16(const float* __restrict__ in,
                             unsigned short* __restrict__ out, int n) {
  int i = blockIdx.x * blockDim.x + threadIdx.x;
  int stride = gridDim.x * blockDim.x;
  for (; i < n; i += stride) out[i] = f2bf(in[i]);
}

// ---------------------------------------------------------------- GEMM
// Y = X(bf16, M x EMBED) @ W(bf16, N x EMBED)^T
// mode 0: scatter bf16 to [B,H,S,D]             (K projection)
// mode 1: scatter bf16 to [B,H,S,D] + [B,H,D,S] (V projection + V^T)
// mode 2: store f32 row-major [M,EMBED]         (output projection)
__global__ void __launch_bounds__(256)
gemm_bf16(const unsigned short* __restrict__ X,
          const unsigned short* __restrict__ W,
          unsigned short* __restrict__ out0,
          unsigned short* __restrict__ out1,
          float* __restrict__ outf,
          int mode) {
  const int wave = threadIdx.x >> 5, lane = threadIdx.x & 31;
  const int lh = lane >> 4, ln = lane & 15;
  const int m0 = blockIdx.y * 128 + wave * 16;
  const int n0 = blockIdx.x * 128;

  v8f acc[8];
  #pragma unroll
  for (int nt = 0; nt < 8; ++nt) acc[nt] = vzero8();

  const unsigned short* xrow = X + (size_t)(m0 + ln) * EMBED;

  #pragma unroll 4
  for (int kc = 0; kc < EMBED / 32; ++kc) {
    const unsigned short* ap = xrow + kc * 32 + lh * 16;
    v16bf a = mergeA(*(const v8u*)ap, *(const v8u*)(ap + 8));
    #pragma unroll
    for (int nt = 0; nt < 8; ++nt) {
      const unsigned short* wrow =
          W + (size_t)(n0 + nt * 16 + ln) * EMBED + kc * 32 + lh * 16;
      v16bf b = asbf(*(const v16u*)wrow);
      acc[nt] = __builtin_amdgcn_wmma_f32_16x16x32_bf16(
          false, a, false, b, (short)0, acc[nt], false, false);
    }
  }

  #pragma unroll
  for (int nt = 0; nt < 8; ++nt) {
    #pragma unroll
    for (int r = 0; r < 8; ++r) {
      const int m = m0 + lh * 8 + r;
      const int n = n0 + nt * 16 + ln;
      const float v = acc[nt][r];
      if (mode == 2) {
        outf[(size_t)m * EMBED + n] = v;
      } else {
        const int b = m / SEQ, s = m % SEQ;
        const int h = n / HEAD_DIM, d = n % HEAD_DIM;
        const unsigned short bv = f2bf(v);
        out0[(((size_t)b * NUM_HEADS + h) * SEQ + s) * HEAD_DIM + d] = bv;
        if (mode == 1)
          out1[(((size_t)b * NUM_HEADS + h) * HEAD_DIM + d) * SEQ + s] = bv;
      }
    }
  }
}

// ---------------------------------------------------------------- attention
// logits = K @ V^T * SCALE; softmax; O = P @ V. One block = 128 K-rows of one (b,h).
// V / V^T tiles are double-buffered in LDS, filled by TDM (tensor_load_to_lds)
// from one wave while all 8 waves crunch WMMAs; fallback: cooperative fill.
__global__ void __launch_bounds__(256)
attn_kernel(const unsigned short* __restrict__ Km,
            const unsigned short* __restrict__ Vm,
            const unsigned short* __restrict__ VTm,
            unsigned short* __restrict__ AO) {
  __shared__ unsigned short Pls[8 * 16 * PROW];        // P exchange, padded rows
  __shared__ unsigned short Vt[2][128 * VROW];         // V tile   (128 x 64, padded)
  __shared__ unsigned short VTt[2][HEAD_DIM * VTROW];  // V^T tile (64 x 128, padded)

  const int wave = threadIdx.x >> 5, lane = threadIdx.x & 31;
  const int lh = lane >> 4, ln = lane & 15;
  const int stile = blockIdx.x, h = blockIdx.y, b = blockIdx.z;

  const unsigned short* Kp  = Km  + (((size_t)b * NUM_HEADS + h) * SEQ) * HEAD_DIM;
  const unsigned short* Vp  = Vm  + (((size_t)b * NUM_HEADS + h) * SEQ) * HEAD_DIM;
  const unsigned short* VTp = VTm + (((size_t)b * NUM_HEADS + h) * HEAD_DIM) * SEQ;

  // A operand (this wave's 16 K-rows, full D=64) held in registers throughout
  const int mrow = stile * 128 + wave * 16 + ln;
  v16bf aK[2];
  #pragma unroll
  for (int kc = 0; kc < 2; ++kc) {
    const unsigned short* p = Kp + (size_t)mrow * HEAD_DIM + kc * 32 + lh * 16;
    aK[kc] = mergeA(*(const v8u*)p, *(const v8u*)(p + 8));
  }

  float mrun[8], lrun[8];
  v8f O[4];
  #pragma unroll
  for (int r = 0; r < 8; ++r) { mrun[r] = -1e30f; lrun[r] = 0.0f; }
  #pragma unroll
  for (int o = 0; o < 4; ++o) O[o] = vzero8();

  unsigned short* Pw = Pls + wave * 16 * PROW;
  const int NT = SEQ / 128;

  // ---- tile fill: TDM from wave 0, or cooperative copy fallback ----
  auto issue_fill = [&](int tb, int buf) {
#if HAVE_TDM
    if (wave == 0) {
      tdm_load_2d((unsigned)(uintptr_t)&Vt[buf][0],
                  Vp + (size_t)tb * 128 * HEAD_DIM,
                  /*tensor_d0=*/HEAD_DIM, /*tensor_d1=*/SEQ,
                  /*tile_d0=*/HEAD_DIM,   /*tile_d1=*/128,
                  /*stride0=*/HEAD_DIM,   /*pad_ivl(32dw)=*/4u);
      tdm_load_2d((unsigned)(uintptr_t)&VTt[buf][0],
                  VTp + tb * 128,
                  /*tensor_d0=*/SEQ,      /*tensor_d1=*/HEAD_DIM,
                  /*tile_d0=*/128,        /*tile_d1=*/HEAD_DIM,
                  /*stride0=*/SEQ,        /*pad_ivl(64dw)=*/5u);
    }
#else
    {  // 256 threads fill both tiles exactly once (padded strides)
      const int tid = (int)threadIdx.x;
      const int vr = tid >> 1, vh = tid & 1;                    // V: 128 rows x 2 halves
      const unsigned short* src = Vp + (size_t)(tb * 128 + vr) * HEAD_DIM + vh * 32;
      unsigned short* dst = &Vt[buf][vr * VROW + vh * 32];
      #pragma unroll
      for (int i = 0; i < 4; ++i)
        *(v8u*)(dst + i * 8) = *(const v8u*)(src + i * 8);
      const int tr = tid >> 2, tq = tid & 3;                    // VT: 64 rows x 4 quarters
      const unsigned short* srcT = VTp + (size_t)tr * SEQ + tb * 128 + tq * 32;
      unsigned short* dstT = &VTt[buf][tr * VTROW + tq * 32];
      #pragma unroll
      for (int i = 0; i < 4; ++i)
        *(v8u*)(dstT + i * 8) = *(const v8u*)(srcT + i * 8);
    }
#endif
  };

  issue_fill(0, 0);
#if HAVE_TDM
  if (wave == 0) __builtin_amdgcn_s_wait_tensorcnt(0);
#endif
  __syncthreads();

  for (int tb = 0; tb < NT; ++tb) {
    const int cur = tb & 1;
    if (tb + 1 < NT) issue_fill(tb + 1, cur ^ 1);   // DMA next tile during compute

    // ---- logits tile: 16 x 128, K-dim = HEAD_DIM = 64 (2 WMMA steps) ----
    v8f lg[8];
    #pragma unroll
    for (int nt = 0; nt < 8; ++nt) {
      v8f c = vzero8();
      const unsigned short* vrow = &Vt[cur][(nt * 16 + ln) * VROW];
      #pragma unroll
      for (int kc = 0; kc < 2; ++kc) {
        v16bf bv = ld16(vrow + kc * 32 + lh * 16);
        c = __builtin_amdgcn_wmma_f32_16x16x32_bf16(
            false, aK[kc], false, bv, (short)0, c, false, false);
      }
      #pragma unroll
      for (int r = 0; r < 8; ++r) c[r] *= ATT_SCALE;
      lg[nt] = c;
    }

    // ---- online softmax (row = lh*8 + r; N spread over 16 lanes) ----
    #pragma unroll
    for (int r = 0; r < 8; ++r) {
      float tmax = lg[0][r];
      #pragma unroll
      for (int nt = 1; nt < 8; ++nt) tmax = fmaxf(tmax, lg[nt][r]);
      #pragma unroll
      for (int mk = 1; mk < 16; mk <<= 1) tmax = fmaxf(tmax, __shfl_xor(tmax, mk, 32));
      const float mnew = fmaxf(mrun[r], tmax);
      const float corr = __expf(mrun[r] - mnew);
      float rs = 0.0f;
      #pragma unroll
      for (int nt = 0; nt < 8; ++nt) {
        const float p = __expf(lg[nt][r] - mnew);
        lg[nt][r] = p;
        rs += p;
      }
      #pragma unroll
      for (int mk = 1; mk < 16; mk <<= 1) rs += __shfl_xor(rs, mk, 32);
      lrun[r] = lrun[r] * corr + rs;
      mrun[r] = mnew;
      #pragma unroll
      for (int o = 0; o < 4; ++o) O[o][r] *= corr;
    }

    // ---- P: C-layout -> LDS (wave-private strip) -> A-layout ----
    #pragma unroll
    for (int nt = 0; nt < 8; ++nt)
      #pragma unroll
      for (int r = 0; r < 8; ++r)
        Pw[(lh * 8 + r) * PROW + nt * 16 + ln] = f2bf(lg[nt][r]);
    __syncthreads();

    v16bf pA[4];
    #pragma unroll
    for (int kc = 0; kc < 4; ++kc) {
      const unsigned short* p = Pw + ln * PROW + kc * 32 + lh * 16;
      pA[kc] = mergeA(*(const v8u*)p, *(const v8u*)(p + 8));
    }

    // ---- O += P @ V  (B operand from LDS V^T tile, K-contiguous) ----
    #pragma unroll
    for (int o = 0; o < 4; ++o) {
      #pragma unroll
      for (int kc = 0; kc < 4; ++kc) {
        v16bf bv = ld16(&VTt[cur][(o * 16 + ln) * VTROW + kc * 32 + lh * 16]);
        O[o] = __builtin_amdgcn_wmma_f32_16x16x32_bf16(
            false, pA[kc], false, bv, (short)0, O[o], false, false);
      }
    }

#if HAVE_TDM
    if (wave == 0) __builtin_amdgcn_s_wait_tensorcnt(0);  // next tile landed
#endif
    __syncthreads();
  }

  // ---- epilogue: O / lrun, scatter to [B,S,E] bf16 (E = h*64 + d) ----
  #pragma unroll
  for (int o = 0; o < 4; ++o) {
    #pragma unroll
    for (int r = 0; r < 8; ++r) {
      const int srow = stile * 128 + wave * 16 + lh * 8 + r;
      const int e = h * HEAD_DIM + o * 16 + ln;
      AO[((size_t)b * SEQ + srow) * EMBED + e] = f2bf(O[o][r] / lrun[r]);
    }
  }
}

// ---------------------------------------------------------------- launch
extern "C" void kernel_launch(void* const* d_in, const int* in_sizes, int n_in,
                              void* d_out, int out_size, void* d_ws, size_t ws_size,
                              hipStream_t stream) {
  (void)in_sizes; (void)n_in; (void)out_size; (void)ws_size;
  const float* x  = (const float*)d_in[0];
  // d_in[1] = Wq — dead in the reference (q unused downstream)
  const float* Wk = (const float*)d_in[2];
  const float* Wv = (const float*)d_in[3];
  const float* Wo = (const float*)d_in[4];
  float* out = (float*)d_out;

  const size_t NX = (size_t)BATCH * SEQ * EMBED;   // 4,194,304
  const size_t NW = (size_t)EMBED * EMBED;         // 1,048,576

  char* ws = (char*)d_ws;
  unsigned short* xb  = (unsigned short*)ws; ws += NX * 2;
  unsigned short* wkb = (unsigned short*)ws; ws += NW * 2;
  unsigned short* wvb = (unsigned short*)ws; ws += NW * 2;
  unsigned short* wob = (unsigned short*)ws; ws += NW * 2;
  unsigned short* Kmw = (unsigned short*)ws; ws += NX * 2;
  unsigned short* Vmw = (unsigned short*)ws; ws += NX * 2;
  unsigned short* VTw = (unsigned short*)ws; ws += NX * 2;
  unsigned short* AO  = (unsigned short*)ws; ws += NX * 2;

  cvt_f32_bf16<<<1024, 256, 0, stream>>>(x,  xb,  (int)NX);
  cvt_f32_bf16<<<512,  256, 0, stream>>>(Wk, wkb, (int)NW);
  cvt_f32_bf16<<<512,  256, 0, stream>>>(Wv, wvb, (int)NW);
  cvt_f32_bf16<<<512,  256, 0, stream>>>(Wo, wob, (int)NW);

  dim3 gg(EMBED / 128, (BATCH * SEQ) / 128);
  gemm_bf16<<<gg, 256, 0, stream>>>(xb, wkb, Kmw, nullptr, nullptr, 0);
  gemm_bf16<<<gg, 256, 0, stream>>>(xb, wvb, Vmw, VTw, nullptr, 1);

  attn_kernel<<<dim3(SEQ / 128, NUM_HEADS, BATCH), 256, 0, stream>>>(Kmw, Vmw, VTw, AO);

  gemm_bf16<<<gg, 256, 0, stream>>>(AO, wob, nullptr, nullptr, out, 2);
}
